// Sot_18451179504178
// MI455X (gfx1250) — matrix operand order repaired and checked by reading, not compiled, not run on hardware
//
#include <hip/hip_runtime.h>
#include <math.h>

#define N 8192
#define KDIM 128
#define LINV 10.0f          // 1/lambda
#define NEG_DIAG -10000.0f  // -DIAG_VAL/lambda

typedef __attribute__((ext_vector_type(4)))  _Float16 h4;
typedef __attribute__((ext_vector_type(8)))  _Float16 h8;
typedef __attribute__((ext_vector_type(16))) _Float16 v16h;
typedef __attribute__((ext_vector_type(8)))  float    v8f;

// ---------------------------------------------------------------------------
// Kernel 1: row-normalize x (f32) -> d (f16).  One wave per row (wave32).
// ---------------------------------------------------------------------------
__global__ void normalize_rows(const float* __restrict__ x, _Float16* __restrict__ d)
{
    const int wave = (blockIdx.x * blockDim.x + threadIdx.x) >> 5;
    const int lane = threadIdx.x & 31;
    const float4 v = ((const float4*)(x + (size_t)wave * KDIM))[lane];
    float ss = v.x * v.x + v.y * v.y + v.z * v.z + v.w * v.w;
#pragma unroll
    for (int m = 16; m >= 1; m >>= 1) ss += __shfl_xor(ss, m, 32);
    const float inv = rsqrtf(ss);
    h4 o;
    o[0] = (_Float16)(v.x * inv);
    o[1] = (_Float16)(v.y * inv);
    o[2] = (_Float16)(v.z * inv);
    o[3] = (_Float16)(v.w * inv);
    ((h4*)(d + (size_t)wave * KDIM))[lane] = o;
}

__global__ void init_logv(float* __restrict__ logv)
{
    logv[blockIdx.x * 256 + threadIdx.x] = 0.0f;
}

// ---------------------------------------------------------------------------
// Kernel 2: log_K = -10*(1 - d @ d^T), diag = -10000, written into d_out.
// One 16x16 output tile per wave via 4x v_wmma_f32_16x16x32_f16 (K=128).
// A-fragment (16x32 f16): lane l (m = l&15, h = l>>4) holds K = {16h..16h+7} and
// {16h+8..16h+15}  -> two contiguous 16B loads from row m.
// B-fragment (32x16 f16) of d^T: lane l (n = l&15, h = l>>4) holds
// K = {16h..16h+15} of column n  == one contiguous 32B load from d row n.
// D layout: VGPR r holds (row = r + 8h, col = l&15).
// ---------------------------------------------------------------------------
__global__ void gemm_logk(const _Float16* __restrict__ d, float* __restrict__ logK)
{
    const int waveId = threadIdx.x >> 5;
    const int lane   = threadIdx.x & 31;
    const int r      = lane & 15;
    const int h      = lane >> 4;

    const int mTile = blockIdx.x >> 6;                    // 512 tile-rows
    const int nTile = ((blockIdx.x & 63) << 3) + waveId;  // 8 tiles per block
    const int m0 = mTile << 4;
    const int n0 = nTile << 4;

    const _Float16* arow = d + (size_t)(m0 + r) * KDIM;
    const _Float16* brow = d + (size_t)(n0 + r) * KDIM;

    v8f acc = {};
#pragma unroll
    for (int c = 0; c < 4; ++c) {
        const int kc = c * 32;
        const h8 alo = *(const h8*)(arow + kc + 16 * h);
        const h8 ahi = *(const h8*)(arow + kc + 16 * h + 8);
        v16h a;
#pragma unroll
        for (int j = 0; j < 8; ++j) { a[j] = alo[j]; a[8 + j] = ahi[j]; }
        const v16h b = *(const v16h*)(brow + kc + 16 * h);
        acc = __builtin_amdgcn_wmma_f32_16x16x32_f16(
            /*neg_a=*/false, a, /*neg_b=*/false, b,
            /*c_mod=*/(short)0, acc, /*reuse_a=*/false, /*reuse_b=*/false);
    }

    const int col = n0 + r;
#pragma unroll
    for (int rr = 0; rr < 8; ++rr) {
        const int row = m0 + rr + 8 * h;
        float val = LINV * acc[rr] - LINV;   // -10*(1 - dot)
        if (row == col) val = NEG_DIAG;
        logK[(size_t)row * N + col] = val;
    }
}

// ---------------------------------------------------------------------------
// Kernel 3: log_u[i] = -LSE_j(log_K[i,j] + log_v[j]).  Block per row,
// log_v staged in LDS, single-pass online LSE + LDS tree reduction.
// ---------------------------------------------------------------------------
__global__ void row_lse(const float* __restrict__ logK, const float* __restrict__ logv,
                        float* __restrict__ logu)
{
    __shared__ float sv[N];
    __shared__ float sm[256], ssum[256];
    const int tid = threadIdx.x;
    const int row = blockIdx.x;

    for (int j = tid; j < N; j += 256) sv[j] = logv[j];
    __syncthreads();

    const float* rowp = logK + (size_t)row * N;
    float m = -INFINITY, s = 0.0f;
    for (int j = tid; j < N; j += 256) {
        const float v = rowp[j] + sv[j];
        if (v > m) { s = s * __expf(m - v) + 1.0f; m = v; }
        else       { s += __expf(v - m); }
    }
    sm[tid] = m; ssum[tid] = s;
    __syncthreads();
    for (int off = 128; off > 0; off >>= 1) {
        if (tid < off) {
            const float m1 = sm[tid], s1 = ssum[tid];
            const float m2 = sm[tid + off], s2 = ssum[tid + off];
            const float M = fmaxf(m1, m2);
            sm[tid]   = M;
            ssum[tid] = s1 * __expf(m1 - M) + s2 * __expf(m2 - M);
        }
        __syncthreads();
    }
    if (tid == 0) logu[row] = -(sm[0] + __logf(ssum[0]));
}

// ---------------------------------------------------------------------------
// Kernel 4a: column-LSE partials.  Block (stripe s = blockIdx.x, group g =
// blockIdx.y): thread owns column j = 256*s + tid, scans rows [256g, 256g+256)
// (coalesced), emits running (max, sum).
// ---------------------------------------------------------------------------
__global__ void col_lse_partial(const float* __restrict__ logK, const float* __restrict__ logu,
                                float* __restrict__ pm, float* __restrict__ ps)
{
    __shared__ float su[256];
    const int tid = threadIdx.x;
    const int j   = blockIdx.x * 256 + tid;
    const int i0  = blockIdx.y * 256;

    su[tid] = logu[i0 + tid];
    __syncthreads();

    const float* p = logK + (size_t)i0 * N + j;
    float m = -INFINITY, s = 0.0f;
    for (int ii = 0; ii < 256; ++ii) {
        const float v = su[ii] + p[(size_t)ii * N];
        if (v > m) { s = s * __expf(m - v) + 1.0f; m = v; }
        else       { s += __expf(v - m); }
    }
    pm[(size_t)blockIdx.y * N + j] = m;
    ps[(size_t)blockIdx.y * N + j] = s;
}

// Kernel 4b: combine 32 partials per column -> log_v[j]
__global__ void col_lse_combine(const float* __restrict__ pm, const float* __restrict__ ps,
                                float* __restrict__ logv)
{
    const int j = blockIdx.x * 256 + threadIdx.x;
    float m = -INFINITY, s = 0.0f;
#pragma unroll
    for (int g = 0; g < 32; ++g) {
        const float m2 = pm[(size_t)g * N + j];
        const float s2 = ps[(size_t)g * N + j];
        const float M  = fmaxf(m, m2);
        s = s * __expf(m - M) + s2 * __expf(m2 - M);
        m = M;
    }
    logv[j] = -(m + __logf(s));
}

// ---------------------------------------------------------------------------
// Kernel 5: in-place P = exp(log_u + log_K + log_v), diag = 1.
// ---------------------------------------------------------------------------
__global__ void finalize(float* __restrict__ P, const float* __restrict__ logu,
                         const float* __restrict__ logv)
{
    __shared__ float sv[N];
    const int tid = threadIdx.x;
    const int row = blockIdx.x;

    for (int j = tid; j < N; j += 256) sv[j] = logv[j];
    __syncthreads();

    const float u = logu[row];
    float* rowp = P + (size_t)row * N;
    for (int j = tid; j < N; j += 256) {
        const float z = __expf(u + rowp[j] + sv[j]);
        rowp[j] = (j == row) ? 1.0f : z;
    }
}

// ---------------------------------------------------------------------------
extern "C" void kernel_launch(void* const* d_in, const int* in_sizes, int n_in,
                              void* d_out, int out_size, void* d_ws, size_t ws_size,
                              hipStream_t stream)
{
    (void)in_sizes; (void)n_in; (void)out_size; (void)ws_size;

    const float* x = (const float*)d_in[0];
    float* out = (float*)d_out;           // log_K lives here, transformed in place
    char* ws = (char*)d_ws;

    _Float16* d16 = (_Float16*)ws;                       // 8192*128*2   = 2 MB
    float* logu   = (float*)(ws + (size_t)(2 << 20));    // 32 KB
    float* logv   = logu + N;                            // 32 KB
    float* pm     = logv + N;                            // 32*8192*4    = 1 MB
    float* ps     = pm + 32 * (size_t)N;                 // 1 MB

    normalize_rows<<<N / 8, 256, 0, stream>>>(x, d16);
    init_logv<<<N / 256, 256, 0, stream>>>(logv);
    gemm_logk<<<(N / 16) * (N / 16) / 8, 256, 0, stream>>>(d16, out);

    for (int it = 0; it < 20; ++it) {
        row_lse<<<N, 256, 0, stream>>>(out, logv, logu);
        col_lse_partial<<<dim3(N / 256, 32), 256, 0, stream>>>(out, logu, pm, ps);
        col_lse_combine<<<N / 256, 256, 0, stream>>>(pm, ps, logv);
    }

    finalize<<<N, 256, 0, stream>>>(out, logu, logv);
}
